// MultiHeadAttention_11716670784070
// MI455X (gfx1250) — matrix operand order repaired and checked by reading, not compiled, not run on hardware
//
#include <hip/hip_runtime.h>
#include <hip/hip_bf16.h>

// ---------------------------------------------------------------------------
// MHA forward for (N=4, L=2048, D=1024, H=16, Dh=64), fp32 in/out.
// All GEMMs on v_wmma_f32_16x16x32_bf16; A-tiles staged to LDS with
// global_load_async_to_lds_b128 (ASYNCcnt) and double-buffered.
// Weights and V are pre-transposed so every WMMA operand loads as b128.
// ---------------------------------------------------------------------------

typedef __attribute__((ext_vector_type(16))) __bf16 bf16x16;
typedef __attribute__((ext_vector_type(8)))  float  f32x8;

#define WMMA_BF16(a, b, c) \
  __builtin_amdgcn_wmma_f32_16x16x32_bf16(false, (a), false, (b), (short)0, (c), false, false)

constexpr int NB     = 4;
constexpr int SEQ    = 2048;
constexpr int DMODEL = 1024;
constexpr int NHEAD  = 16;
constexpr int DHEAD  = 64;
constexpr int QBLK   = 64;                 // query rows per wave in flash attn
constexpr int NQB    = SEQ / QBLK;         // 32
constexpr int ROWS   = NB * SEQ;           // 8192 (GEMM M)

// ---------------------------------------------------------------------------
// CDNA5 async copy: per-lane 16B global -> LDS, tracked by ASYNCcnt.
// ---------------------------------------------------------------------------
__device__ __forceinline__ void async_copy_b128(unsigned lds_off, const void* gaddr) {
  asm volatile("global_load_async_to_lds_b128 %0, %1, off"
               :: "v"(lds_off), "v"(gaddr)
               : "memory");
}
__device__ __forceinline__ void wait_async_le4() {
  asm volatile("s_wait_asynccnt 0x4" ::: "memory");
}
__device__ __forceinline__ void wait_async_0() {
  asm volatile("s_wait_asynccnt 0x0" ::: "memory");
}

// ---------------------------------------------------------------------------
// WMMA fragment loaders (wave32; contiguous k = 16*half + s for both A and B)
// ---------------------------------------------------------------------------

// A fragment, 16x32 bf16, source row-major with leading dim `ld`.
__device__ __forceinline__ bf16x16 load_a16x32(const __bf16* __restrict__ p, int ld, int lane) {
  const int half = lane >> 4, r = lane & 15;
  const __bf16* rowp = p + (size_t)r * ld;
  bf16x16 a;
#pragma unroll
  for (int i = 0; i < 8; ++i) {
    const int k = 16 * half + 2 * i;
    a[2 * i]     = rowp[k];
    a[2 * i + 1] = rowp[k + 1];
  }
  return a;
}

// B fragment (32x16) where the logical B is the TRANSPOSE of a row-major
// source: B[k][n] = p[n*ld + k].  Per-lane reads are 16 contiguous bf16.
__device__ __forceinline__ bf16x16 load_bT32x16(const __bf16* __restrict__ p, int ld, int lane) {
  const int half = lane >> 4, n = lane & 15;
  const __bf16* rowp = p + (size_t)n * ld;
  bf16x16 b;
#pragma unroll
  for (int i = 0; i < 8; ++i) {
    const int k = 16 * half + 2 * i;
    b[2 * i]     = rowp[k];
    b[2 * i + 1] = rowp[k + 1];
  }
  return b;
}

// ---------------------------------------------------------------------------
// fp32 -> bf16 conversion (straight)
// ---------------------------------------------------------------------------
__global__ void cvt_f32_bf16(const float* __restrict__ in, __bf16* __restrict__ out, int n) {
  int i = blockIdx.x * blockDim.x + threadIdx.x;
  if (i < n) out[i] = (__bf16)in[i];
}

// fp32 [R x C] row-major -> bf16 [C x R] row-major (transpose), LDS-tiled.
__global__ __launch_bounds__(256) void cvt_transpose_bf16(
    const float* __restrict__ in, __bf16* __restrict__ out, int R, int C) {
  __shared__ float tile[32][33];
  const int c0 = blockIdx.x * 32, r0 = blockIdx.y * 32;
  const int tx = threadIdx.x, ty = threadIdx.y;   // block (32, 8)
#pragma unroll
  for (int yy = ty; yy < 32; yy += 8)
    tile[yy][tx] = in[(size_t)(r0 + yy) * C + (c0 + tx)];
  __syncthreads();
#pragma unroll
  for (int yy = ty; yy < 32; yy += 8)
    out[(size_t)(c0 + yy) * R + (r0 + tx)] = (__bf16)tile[tx][yy];
}

// ---------------------------------------------------------------------------
// GEMM + bias: C[M,N] (f32) = A[M,K] (bf16) @ Bt[N,K]^T (bf16) + bias[N]
// Workgroup = 4 waves, 128x128 tile; each wave computes 64x64 via 4x4 WMMAs.
// A slab (128x32) is double-buffered in LDS via async copies; Bt fragments
// are per-lane contiguous b128 loads from global (L2-resident weights).
// ---------------------------------------------------------------------------
__global__ __launch_bounds__(128) void gemm_bias_kernel(
    const __bf16* __restrict__ A, const __bf16* __restrict__ Bt,
    const float* __restrict__ bias, float* __restrict__ C,
    int M, int N, int K) {
  __shared__ __bf16 Atile[2][128][32];      // 8 KB per buffer

  const int lane = threadIdx.x & 31;
  const int wave = threadIdx.x >> 5;
  const int m0   = blockIdx.x * 128;
  const int mw   = (wave >> 1) * 64;        // wave's row offset inside tile
  const int n0   = blockIdx.y * 128 + (wave & 1) * 64;

  // stage one 128x32 A-slab: thread t copies row t (64B) as 4 async b128s
  auto stage = [&](int buf, int k0) {
    const __bf16* src = A + (size_t)(m0 + threadIdx.x) * K + k0;
    const unsigned dst = (unsigned)(size_t)&Atile[buf][threadIdx.x][0];
    async_copy_b128(dst,      src);
    async_copy_b128(dst + 16, src + 8);
    async_copy_b128(dst + 32, src + 16);
    async_copy_b128(dst + 48, src + 24);
  };

  f32x8 acc[4][4];
#pragma unroll
  for (int i = 0; i < 4; ++i)
#pragma unroll
    for (int j = 0; j < 4; ++j)
#pragma unroll
      for (int e = 0; e < 8; ++e) acc[i][j][e] = 0.0f;

  stage(0, 0);
  int buf = 0;
  for (int k0 = 0; k0 < K; k0 += 32) {
    if (k0 + 32 < K) {
      stage(buf ^ 1, k0 + 32);
      __builtin_prefetch(Bt + (size_t)n0 * K + (k0 + 32), 0, 3);
      wait_async_le4();            // drain current buffer (in-order completion)
    } else {
      wait_async_0();
    }
    __syncthreads();               // slab visible to all 4 waves

    bf16x16 af[4], bfr[4];
#pragma unroll
    for (int i = 0; i < 4; ++i)
      af[i] = load_a16x32(&Atile[buf][mw + 16 * i][0], 32, lane);
#pragma unroll
    for (int j = 0; j < 4; ++j)
      bfr[j] = load_bT32x16(Bt + (size_t)(n0 + 16 * j) * K + k0, K, lane);
#pragma unroll
    for (int i = 0; i < 4; ++i)
#pragma unroll
      for (int j = 0; j < 4; ++j)
        acc[i][j] = WMMA_BF16(af[i], bfr[j], acc[i][j]);

    __syncthreads();               // all waves done reading before re-stage
    buf ^= 1;
  }

  const int half = lane >> 4, nn = lane & 15;
#pragma unroll
  for (int j = 0; j < 4; ++j) {
    const int col = n0 + j * 16 + nn;
    const float bv = bias[col];
#pragma unroll
    for (int i = 0; i < 4; ++i)
#pragma unroll
      for (int r = 0; r < 8; ++r) {
        const int row = m0 + mw + i * 16 + r + 8 * half;
        C[(size_t)row * N + col] = acc[i][j][r] + bv;
      }
  }
}

// ---------------------------------------------------------------------------
// RoPE on q,k + split/cast to bf16.  Q,K head-major [N,H,L,Dh]; V stored
// TRANSPOSED as [N,H,Dh,L] so attention's P@V operand loads contiguously.
// One thread per (b, l, h, i), i in [0,32) handles rotation pair (i, i+32).
// ---------------------------------------------------------------------------
__global__ void rope_split_kernel(const float* __restrict__ qkv,
                                  const int* __restrict__ tids,
                                  __bf16* __restrict__ Q, __bf16* __restrict__ K,
                                  __bf16* __restrict__ Vt) {
  int t = blockIdx.x * blockDim.x + threadIdx.x;
  const int i = t & 31;        t >>= 5;
  const int h = t & (NHEAD-1); t >>= 4;
  const int l = t & (SEQ-1);   t >>= 11;
  const int b = t;
  if (b >= NB) return;

  const float tid  = (float)tids[b * SEQ + l];
  const float freq = __powf(10000.0f, -(float)(2 * i) / (float)DHEAD);
  float sn, cs;
  __sincosf(tid * freq, &sn, &cs);

  const size_t rowoff = ((size_t)b * SEQ + l) * (3 * DMODEL);
  const int f1 = h * DHEAD + i;
  const int f2 = h * DHEAD + 32 + i;

  const float q1 = qkv[rowoff + f1],              q2 = qkv[rowoff + f2];
  const float k1 = qkv[rowoff + DMODEL + f1],     k2 = qkv[rowoff + DMODEL + f2];
  const float v1 = qkv[rowoff + 2 * DMODEL + f1], v2 = qkv[rowoff + 2 * DMODEL + f2];

  const size_t ho = (((size_t)(b * NHEAD + h)) * SEQ + l) * DHEAD;
  Q[ho + i]      = (__bf16)(q1 * cs - q2 * sn);
  Q[ho + 32 + i] = (__bf16)(q1 * sn + q2 * cs);
  K[ho + i]      = (__bf16)(k1 * cs - k2 * sn);
  K[ho + 32 + i] = (__bf16)(k1 * sn + k2 * cs);

  const size_t vbase = ((size_t)(b * NHEAD + h)) * DHEAD * SEQ;   // [Dh][L]
  Vt[vbase + (size_t)i        * SEQ + l] = (__bf16)v1;
  Vt[vbase + (size_t)(32 + i) * SEQ + l] = (__bf16)v2;
}

// ---------------------------------------------------------------------------
// Flash attention: one wave per (b, h, 64-query block).
// Online softmax over 64-key blocks; S and PV both on WMMA.
// K is [N,H,L,Dh] (Q@K^T operand is naturally contiguous);
// V is [N,H,Dh,L] (P@V operand is contiguous).
// Output written as [N, L, D] bf16 (A matrix of the final GEMM).
// ---------------------------------------------------------------------------
__global__ __launch_bounds__(32) void flash_attn_kernel(
    const __bf16* __restrict__ Q, const __bf16* __restrict__ K,
    const __bf16* __restrict__ Vt, const unsigned char* __restrict__ mask,
    __bf16* __restrict__ O) {
  int idx = blockIdx.x;
  const int qb = idx % NQB; idx /= NQB;
  const int h  = idx % NHEAD; idx /= NHEAD;
  const int b  = idx;

  const int lane = threadIdx.x;
  const int half = lane >> 4, nn = lane & 15;
  const int q0 = qb * QBLK;

  const size_t bh = (size_t)(b * NHEAD + h) * SEQ * DHEAD;
  const __bf16* Qbh = Q  + bh;
  const __bf16* Kbh = K  + bh;
  const __bf16* Vbh = Vt + bh;                 // [Dh][SEQ]
  const unsigned char* maskrow = mask + (size_t)b * SEQ;

  __shared__ __bf16 Plds[QBLK][QBLK + 8];      // C-layout -> A-layout staging

  bf16x16 qf[4][2];
#pragma unroll
  for (int i = 0; i < 4; ++i)
#pragma unroll
    for (int kt = 0; kt < 2; ++kt)
      qf[i][kt] = load_a16x32(Qbh + (size_t)(q0 + 16 * i) * DHEAD + kt * 32, DHEAD, lane);

  f32x8 o[4][4];
  float mrun[4][8], lrun[4][8];
#pragma unroll
  for (int i = 0; i < 4; ++i) {
#pragma unroll
    for (int r = 0; r < 8; ++r) { mrun[i][r] = -1.0e30f; lrun[i][r] = 0.0f; }
#pragma unroll
    for (int j = 0; j < 4; ++j)
#pragma unroll
      for (int e = 0; e < 8; ++e) o[i][j][e] = 0.0f;
  }

  const float scale = 0.125f;                  // 64^-0.5

  for (int kb = 0; kb < SEQ; kb += QBLK) {
    // ---- S = Q @ K^T (64x64, f32) ----
    bf16x16 ktf[4][2];
#pragma unroll
    for (int j = 0; j < 4; ++j)
#pragma unroll
      for (int kt = 0; kt < 2; ++kt)
        ktf[j][kt] = load_bT32x16(Kbh + (size_t)(kb + 16 * j) * DHEAD + kt * 32, DHEAD, lane);

    f32x8 s[4][4];
#pragma unroll
    for (int i = 0; i < 4; ++i)
#pragma unroll
      for (int j = 0; j < 4; ++j) {
#pragma unroll
        for (int e = 0; e < 8; ++e) s[i][j][e] = 0.0f;
        s[i][j] = WMMA_BF16(qf[i][0], ktf[j][0], s[i][j]);
        s[i][j] = WMMA_BF16(qf[i][1], ktf[j][1], s[i][j]);
      }

    // ---- scale + mask bias ----
#pragma unroll
    for (int j = 0; j < 4; ++j) {
      const int key = kb + j * 16 + nn;
      const float mb = maskrow[key] ? 0.0f : -1.0e30f;
#pragma unroll
      for (int i = 0; i < 4; ++i)
#pragma unroll
        for (int r = 0; r < 8; ++r)
          s[i][j][r] = s[i][j][r] * scale + mb;
    }

    // ---- online softmax: row stats across 16-lane half-groups (wave32) ----
#pragma unroll
    for (int i = 0; i < 4; ++i) {
#pragma unroll
      for (int r = 0; r < 8; ++r) {
        float v = fmaxf(fmaxf(s[i][0][r], s[i][1][r]), fmaxf(s[i][2][r], s[i][3][r]));
#pragma unroll
        for (int d = 1; d < 16; d <<= 1) v = fmaxf(v, __shfl_xor(v, d, 16));
        const float mnew  = fmaxf(mrun[i][r], v);
        const float alpha = __expf(mrun[i][r] - mnew);
        float rs = 0.0f;
#pragma unroll
        for (int j = 0; j < 4; ++j) {
          const float p = __expf(s[i][j][r] - mnew);
          s[i][j][r] = p;
          rs += p;
        }
#pragma unroll
        for (int d = 1; d < 16; d <<= 1) rs += __shfl_xor(rs, d, 16);
        lrun[i][r] = lrun[i][r] * alpha + rs;
        mrun[i][r] = mnew;
#pragma unroll
        for (int j = 0; j < 4; ++j) o[i][j][r] *= alpha;
      }
    }

    // ---- P: C-layout -> LDS -> A-layout bf16 fragments ----
#pragma unroll
    for (int i = 0; i < 4; ++i)
#pragma unroll
      for (int j = 0; j < 4; ++j)
#pragma unroll
        for (int r = 0; r < 8; ++r)
          Plds[i * 16 + r + 8 * half][j * 16 + nn] = (__bf16)s[i][j][r];
    // single wave: LDS ops are in-order; compiler inserts waits as needed

    bf16x16 pf[4][2], vf[2][4];
#pragma unroll
    for (int i = 0; i < 4; ++i)
#pragma unroll
      for (int kt = 0; kt < 2; ++kt)
        pf[i][kt] = load_a16x32(&Plds[i * 16][kt * 32], QBLK + 8, lane);
#pragma unroll
    for (int kt = 0; kt < 2; ++kt)
#pragma unroll
      for (int j = 0; j < 4; ++j)   // B[k][n] = V[kb+kt*32+k][j*16+n] = Vbh[(j*16+n)*SEQ + ...]
        vf[kt][j] = load_bT32x16(Vbh + (size_t)(16 * j) * SEQ + kb + kt * 32, SEQ, lane);

    // ---- O += P @ V ----
#pragma unroll
    for (int i = 0; i < 4; ++i)
#pragma unroll
      for (int j = 0; j < 4; ++j) {
        o[i][j] = WMMA_BF16(pf[i][0], vf[0][j], o[i][j]);
        o[i][j] = WMMA_BF16(pf[i][1], vf[1][j], o[i][j]);
      }
  }

  // ---- normalize + store as [N, L, D] bf16 ----
#pragma unroll
  for (int i = 0; i < 4; ++i)
#pragma unroll
    for (int r = 0; r < 8; ++r) {
      const float inv = 1.0f / lrun[i][r];
      const int row = q0 + i * 16 + r + 8 * half;
#pragma unroll
      for (int j = 0; j < 4; ++j) {
        const int col = j * 16 + nn;
        O[((size_t)b * SEQ + row) * DMODEL + h * DHEAD + col] = (__bf16)(o[i][j][r] * inv);
      }
    }
}

// ---------------------------------------------------------------------------
// Launch.  inputs: 0:x 1:mask 2:time_ids 3:W_in 4:b_in 5:W_out 6:b_out
// ---------------------------------------------------------------------------
extern "C" void kernel_launch(void* const* d_in, const int* in_sizes, int n_in,
                              void* d_out, int out_size, void* d_ws, size_t ws_size,
                              hipStream_t stream) {
  const float*         x     = (const float*)d_in[0];
  const unsigned char* mask  = (const unsigned char*)d_in[1];
  const int*           tids  = (const int*)d_in[2];
  const float*         W_in  = (const float*)d_in[3];
  const float*         b_in  = (const float*)d_in[4];
  const float*         W_out = (const float*)d_in[5];
  const float*         b_out = (const float*)d_in[6];
  float*               out   = (float*)d_out;

  size_t off = 0;
  auto take = [&](size_t bytes) -> void* {
    void* p = (char*)d_ws + off;
    off += (bytes + 255) & ~(size_t)255;
    return p;
  };
  __bf16* Xb    = (__bf16*)take((size_t)ROWS * DMODEL * 2);          // 16 MB
  __bf16* Wtin  = (__bf16*)take((size_t)3 * DMODEL * DMODEL * 2);    //  6 MB  [3D x D]
  __bf16* Wtout = (__bf16*)take((size_t)DMODEL * DMODEL * 2);        //  2 MB  [D x D]
  float*  qkv   = (float*) take((size_t)ROWS * 3 * DMODEL * 4);      // 96 MB
  __bf16* Qb    = (__bf16*)take((size_t)ROWS * DMODEL * 2);          // 16 MB
  __bf16* Kb    = (__bf16*)take((size_t)ROWS * DMODEL * 2);          // 16 MB
  __bf16* Vtb   = (__bf16*)take((size_t)ROWS * DMODEL * 2);          // 16 MB
  __bf16* Ob    = Xb;  // Xb dead after QKV GEMM; reuse for attention output

  // 1) conversions: x straight; weights transposed to [N x K]
  {
    int n = ROWS * DMODEL;
    cvt_f32_bf16<<<(n + 255) / 256, 256, 0, stream>>>(x, Xb, n);
    cvt_transpose_bf16<<<dim3((3 * DMODEL) / 32, DMODEL / 32), dim3(32, 8), 0, stream>>>(
        W_in, Wtin, DMODEL, 3 * DMODEL);
    cvt_transpose_bf16<<<dim3(DMODEL / 32, DMODEL / 32), dim3(32, 8), 0, stream>>>(
        W_out, Wtout, DMODEL, DMODEL);
  }

  // 2) QKV projection: [8192,1024] @ [1024,3072] + b_in
  gemm_bias_kernel<<<dim3(ROWS / 128, (3 * DMODEL) / 128), 128, 0, stream>>>(
      Xb, Wtin, b_in, qkv, ROWS, 3 * DMODEL, DMODEL);

  // 3) RoPE + head split (Q,K head-major; V transposed)
  {
    int nthr = NB * SEQ * NHEAD * 32;
    rope_split_kernel<<<nthr / 256, 256, 0, stream>>>(qkv, tids, Qb, Kb, Vtb);
  }

  // 4) Flash attention -> Ob [N,L,D] bf16
  flash_attn_kernel<<<NB * NHEAD * NQB, 32, 0, stream>>>(Qb, Kb, Vtb, mask, Ob);

  // 5) Output projection: [8192,1024] @ [1024,1024] + b_out -> d_out (f32)
  gemm_bias_kernel<<<dim3(ROWS / 128, DMODEL / 128), 128, 0, stream>>>(
      Ob, Wtout, b_out, out, ROWS, DMODEL, DMODEL);

  (void)in_sizes; (void)n_in; (void)out_size; (void)ws_size;
}
